// DeepSpeedMoEInference_76270029243059
// MI455X (gfx1250) — matrix operand order repaired and checked by reading, not compiled, not run on hardware
//
#include <hip/hip_runtime.h>
#include <stdint.h>

// Problem constants (from reference)
#define HIDDEN  1024
#define INTER   4096
#define EXPERTS 8
#define S_TOK   8192          // B*T = 4*2048
#define CAP     1024          // max(4, ceil(S/E * 1.0))

typedef __attribute__((ext_vector_type(16))) __bf16 v16bf;
typedef __attribute__((ext_vector_type(8)))  float  v8f;
typedef int v4i_vs __attribute__((vector_size(16)));   // matches builtin proto

union Frag { uint4 u[2]; v16bf v; };

__device__ __forceinline__ unsigned short f2bf(float f) {
  unsigned u = __float_as_uint(f);
  u += 0x7FFFu + ((u >> 16) & 1u);   // round-to-nearest-even
  return (unsigned short)(u >> 16);
}

__device__ __forceinline__ float gelu_tanh(float x) {
  float x3 = x * x * x;
  float t  = tanhf(0.7978845608028654f * (x + 0.044715f * x3));
  return 0.5f * x * (1.0f + t);
}

// ---- CDNA5 async global->LDS copy (ASYNCcnt path) ----
__device__ __forceinline__ void async_copy_b128(const void* g, void* l) {
#if __has_builtin(__builtin_amdgcn_global_load_async_to_lds_b128)
  __builtin_amdgcn_global_load_async_to_lds_b128(
      (__attribute__((address_space(1))) v4i_vs*)g,
      (__attribute__((address_space(3))) v4i_vs*)l, 0, 0);
#else
  unsigned lo = (unsigned)(unsigned long long)l;  // LDS aperture: low 32 bits
  asm volatile("global_load_async_to_lds_b128 %0, %1, off"
               :: "v"(lo), "v"(g) : "memory");
#endif
}

__device__ __forceinline__ void wait_async0() {
#if __has_builtin(__builtin_amdgcn_s_wait_asynccnt)
  __builtin_amdgcn_s_wait_asynccnt(0);
#else
  asm volatile("s_wait_asynccnt 0x0" ::: "memory");
#endif
}

// ---------------------------------------------------------------------------
// K1: LayerNorm + gate logits (8) + res_coef logits (2), softmax/argmax.
// One block per token, 256 threads, 4 elems/thread.
// ---------------------------------------------------------------------------
__global__ __launch_bounds__(256)
void ln_gate_kernel(const float* __restrict__ x,
                    const float* __restrict__ nw,
                    const float* __restrict__ nb,
                    const float* __restrict__ wg,        // [HIDDEN, EXPERTS]
                    const float* __restrict__ res_coef,  // [HIDDEN, 2]
                    unsigned short* __restrict__ attn_bf,
                    int*   __restrict__ expert_id,
                    float* __restrict__ gatep,
                    float* __restrict__ coef0,
                    float* __restrict__ coef1) {
  const int s   = blockIdx.x;
  const int tid = threadIdx.x;
  __shared__ float r1[256], r2[256];
  __shared__ float red[10][256];

  const float* xs = x + (size_t)s * HIDDEN;
  float v[4];
  float sum = 0.f, sq = 0.f;
#pragma unroll
  for (int j = 0; j < 4; ++j) {
    int k = tid + 256 * j;
    float t = xs[k];
    v[j] = t; sum += t; sq += t * t;
  }
  r1[tid] = sum; r2[tid] = sq;
  __syncthreads();
  for (int st = 128; st > 0; st >>= 1) {
    if (tid < st) { r1[tid] += r1[tid + st]; r2[tid] += r2[tid + st]; }
    __syncthreads();
  }
  const float mean = r1[0] * (1.f / HIDDEN);
  const float var  = r2[0] * (1.f / HIDDEN) - mean * mean;
  const float rstd = rsqrtf(var + 1e-12f);

  float part[10];
#pragma unroll
  for (int a = 0; a < 10; ++a) part[a] = 0.f;
#pragma unroll
  for (int j = 0; j < 4; ++j) {
    int k = tid + 256 * j;
    float nx = (v[j] - mean) * rstd * nw[k] + nb[k];
    attn_bf[(size_t)s * HIDDEN + k] = f2bf(nx);
#pragma unroll
    for (int e = 0; e < EXPERTS; ++e) part[e] += nx * wg[k * EXPERTS + e];
    part[8] += nx * res_coef[k * 2 + 0];
    part[9] += nx * res_coef[k * 2 + 1];
  }
#pragma unroll
  for (int a = 0; a < 10; ++a) red[a][tid] = part[a];
  __syncthreads();
  for (int st = 128; st > 0; st >>= 1) {
    if (tid < st) {
#pragma unroll
      for (int a = 0; a < 10; ++a) red[a][tid] += red[a][tid + st];
    }
    __syncthreads();
  }
  if (tid == 0) {
    float m = red[0][0]; int am = 0;
#pragma unroll
    for (int e = 1; e < EXPERTS; ++e)
      if (red[e][0] > m) { m = red[e][0]; am = e; }
    float den = 0.f;
#pragma unroll
    for (int e = 0; e < EXPERTS; ++e) den += __expf(red[e][0] - m);
    expert_id[s] = am;
    gatep[s]     = 1.f / den;            // softmax prob of argmax expert
    float a = red[8][0], b = red[9][0];
    float mx = fmaxf(a, b);
    float ea = __expf(a - mx), eb = __expf(b - mx);
    float inv = 1.f / (ea + eb);
    coef0[s] = ea * inv;
    coef1[s] = eb * inv;
  }
}

// ---------------------------------------------------------------------------
// K2: deterministic token-order exclusive cumsum per expert -> capacity slots.
// Single block, 1024 threads, 8 consecutive tokens each (order preserved).
// ---------------------------------------------------------------------------
__global__ __launch_bounds__(1024)
void gate_scan_kernel(const int* __restrict__ expert_id,
                      int* __restrict__ slot,
                      int* __restrict__ tos) {   // token-of-slot [E*CAP]
  const int tid = threadIdx.x;
  __shared__ int scn[1024 * EXPERTS];   // 32 KB

  int cnt[EXPERTS];
#pragma unroll
  for (int e = 0; e < EXPERTS; ++e) cnt[e] = 0;
  int eid[8];
#pragma unroll
  for (int r = 0; r < 8; ++r) {
    int s = tid * 8 + r;
    int e = expert_id[s];
    eid[r] = e;
    cnt[e]++;
    tos[s] = -1;                        // init token-of-slot (S_TOK == E*CAP)
  }
#pragma unroll
  for (int e = 0; e < EXPERTS; ++e) scn[tid * EXPERTS + e] = cnt[e];
  __threadfence_block();
  __syncthreads();
  for (int off = 1; off < 1024; off <<= 1) {
    int t8[EXPERTS];
    const bool act = (tid >= off);
    if (act) {
#pragma unroll
      for (int e = 0; e < EXPERTS; ++e) t8[e] = scn[(tid - off) * EXPERTS + e];
    }
    __syncthreads();
    if (act) {
#pragma unroll
      for (int e = 0; e < EXPERTS; ++e) scn[tid * EXPERTS + e] += t8[e];
    }
    __syncthreads();
  }
  int pref[EXPERTS];
#pragma unroll
  for (int e = 0; e < EXPERTS; ++e) pref[e] = scn[tid * EXPERTS + e] - cnt[e];
#pragma unroll
  for (int r = 0; r < 8; ++r) {
    int s = tid * 8 + r;
    int e = eid[r];
    int p = pref[e]++;
    if (p < CAP) { slot[s] = p; tos[e * CAP + p] = s; }
    else         { slot[s] = -1; }
  }
}

// ---------------------------------------------------------------------------
// K3: build dispatched [E, CAP, HIDDEN] bf16 (zero empty slots).
// One block per (e,c) slot, 128 threads x 8 bf16.
// ---------------------------------------------------------------------------
__global__ __launch_bounds__(128)
void dispatch_kernel(const int* __restrict__ tos,
                     const unsigned short* __restrict__ attn_bf,
                     unsigned short* __restrict__ dispA) {
  const int p   = blockIdx.x;           // e*CAP + c
  const int tid = threadIdx.x;
  const int s   = tos[p];
  uint4* dst = (uint4*)(dispA + (size_t)p * HIDDEN);
  if (s >= 0) {
    const uint4* src = (const uint4*)(attn_bf + (size_t)s * HIDDEN);
    dst[tid] = src[tid];
  } else {
    dst[tid] = make_uint4(0u, 0u, 0u, 0u);
  }
}

// ---------------------------------------------------------------------------
// K4/K5: tiled bf16 WMMA GEMM:  out = act(A[MxK]bf16 @ W[KxN]fp32->bf16 + bias)
// 128x128x32 tiles, 256 threads (8 waves), wave = 64x32 region (4x2 WMMA).
// A tiles: async global->LDS (ASYNCcnt).  W tiles: fp32 -> bf16 in registers.
// Double-buffered LDS, one barrier per K-step.
// ---------------------------------------------------------------------------
#define LDS_STRIDE 20   // 16 dwords (32 bf16) + 4 dword pad, 80B (16B aligned)

__device__ __forceinline__ void load_b_tile(const float* __restrict__ W, int N,
                                            int k0, int bk, int col0, int bn,
                                            float wv[4][4]) {
#pragma unroll
  for (int kk = 0; kk < 4; ++kk) {
    float4 t = *(const float4*)(W + (size_t)(k0 + bk + kk) * N + col0 + bn);
    wv[kk][0] = t.x; wv[kk][1] = t.y; wv[kk][2] = t.z; wv[kk][3] = t.w;
  }
}

__device__ __forceinline__ void store_b_tile(unsigned int* Bs, int bn, int bk,
                                             const float wv[4][4]) {
#pragma unroll
  for (int nn = 0; nn < 4; ++nn) {   // transpose 4x4, fp32 -> bf16 pairs
    uint2 pk;
    pk.x = (unsigned)f2bf(wv[0][nn]) | ((unsigned)f2bf(wv[1][nn]) << 16);
    pk.y = (unsigned)f2bf(wv[2][nn]) | ((unsigned)f2bf(wv[3][nn]) << 16);
    *(uint2*)&Bs[(bn + nn) * LDS_STRIDE + (bk >> 1)] = pk;
  }
}

template <bool GELU, bool OUT_BF16>
__global__ __launch_bounds__(256)
void gemm_kernel(const unsigned short* __restrict__ A, size_t aBatch,
                 const float* __restrict__ W, size_t wBatch,
                 const float* __restrict__ bias, size_t bBatch,
                 void* __restrict__ outv, size_t oBatch,
                 int N, int K) {
  const int z = blockIdx.z;
  A    += (size_t)z * aBatch;
  W    += (size_t)z * wBatch;
  bias += (size_t)z * bBatch;

  const int row0 = blockIdx.x * 128;
  const int col0 = blockIdx.y * 128;
  const int tid  = threadIdx.x;
  const int lid  = tid & 31;
  const int wid  = tid >> 5;
  const int wm   = wid >> 2;        // 0..1  -> 64-row slab
  const int wn   = wid & 3;         // 0..3  -> 32-col slab
  const int l16  = lid & 15;
  const int lh   = lid >> 4;        // lane half

  __shared__ unsigned int As[2][128 * LDS_STRIDE];  // [m][k] bf16 pairs
  __shared__ unsigned int Bs[2][128 * LDS_STRIDE];  // [n][k] bf16 pairs (W^T)

  v8f acc[4][2];
#pragma unroll
  for (int i = 0; i < 4; ++i)
#pragma unroll
    for (int j = 0; j < 2; ++j)
#pragma unroll
      for (int r = 0; r < 8; ++r) acc[i][j][r] = 0.f;

  const int arow  = tid >> 1;       // 128 rows, 2 threads per row
  const int ahalf = tid & 1;        // k-half (16 bf16 = 32B each)
  const int bn    = (tid & 31) * 4; // 4 consecutive n per thread
  const int bk    = (tid >> 5) * 4; // 4 consecutive k per thread

  const int nIter = K >> 5;
  float wv[4][4];

  // ---- prologue: tile 0 into buffer 0 ----
  {
    const char* g = (const char*)(A + (size_t)(row0 + arow) * K + 0 + ahalf * 16);
    unsigned int* dst = &As[0][arow * LDS_STRIDE + ahalf * 8];
    async_copy_b128(g, dst);
    async_copy_b128(g + 16, dst + 4);
    load_b_tile(W, N, 0, bk, col0, bn, wv);
    store_b_tile(Bs[0], bn, bk, wv);
    wait_async0();
  }
  __syncthreads();

  for (int it = 0; it < nIter; ++it) {
    const int cur  = it & 1;
    const bool more = (it + 1) < nIter;
    if (more) {
      const int k0n = (it + 1) << 5;
      const char* g = (const char*)(A + (size_t)(row0 + arow) * K + k0n + ahalf * 16);
      unsigned int* dst = &As[cur ^ 1][arow * LDS_STRIDE + ahalf * 8];
      async_copy_b128(g, dst);
      async_copy_b128(g + 16, dst + 4);
      load_b_tile(W, N, k0n, bk, col0, bn, wv);
      store_b_tile(Bs[cur ^ 1], bn, bk, wv);
      if (it + 2 < nIter)   // CDNA5 global_prefetch_b8 for the tile after next
        __builtin_prefetch(W + (size_t)(((it + 2) << 5) + bk) * N + col0 + bn, 0, 1);
    }

    // ---- fragments per ISA 16-bit A/B layouts (from buffer `cur`) ----
    Frag af[4], bfr[2];
#pragma unroll
    for (int i = 0; i < 4; ++i) {
      const int r = wm * 64 + i * 16 + l16;
      af[i].u[0] = *(const uint4*)&As[cur][r * LDS_STRIDE + lh * 4];      // K = lh*8 + 0..7
      af[i].u[1] = *(const uint4*)&As[cur][r * LDS_STRIDE + 8 + lh * 4];  // K = 16 + lh*8 + 0..7
    }
#pragma unroll
    for (int j = 0; j < 2; ++j) {
      const int r = wn * 32 + j * 16 + l16;
      bfr[j].u[0] = *(const uint4*)&Bs[cur][r * LDS_STRIDE + lh * 4];     // K = lh*8 + 0..7
      bfr[j].u[1] = *(const uint4*)&Bs[cur][r * LDS_STRIDE + 8 + lh * 4]; // K = 16 + lh*8 + 0..7
    }
#pragma unroll
    for (int i = 0; i < 4; ++i)
#pragma unroll
      for (int j = 0; j < 2; ++j)
        acc[i][j] = __builtin_amdgcn_wmma_f32_16x16x32_bf16(
            false, af[i].v, false, bfr[j].v, (short)0, acc[i][j], false, false);

    if (more) {
      wait_async0();      // our async writes to buffer cur^1 have landed
      __syncthreads();    // everyone's writes visible, everyone done reading cur
    }
  }

  // ---- epilogue: bias (+gelu), store ----
#pragma unroll
  for (int i = 0; i < 4; ++i)
#pragma unroll
    for (int j = 0; j < 2; ++j) {
      const int n  = col0 + wn * 32 + j * 16 + l16;
      const float bv = bias[n];
#pragma unroll
      for (int r = 0; r < 8; ++r) {
        const int m = row0 + wm * 64 + i * 16 + r + 8 * lh;
        float val = acc[i][j][r] + bv;
        if constexpr (GELU) val = gelu_tanh(val);
        if constexpr (OUT_BF16) {
          unsigned short* out = (unsigned short*)outv + (size_t)z * oBatch;
          out[(size_t)m * N + n] = f2bf(val);
        } else {
          float* out = (float*)outv + (size_t)z * oBatch;
          out[(size_t)m * N + n] = val;
        }
      }
    }
}

// ---------------------------------------------------------------------------
// K6: out = residual + coef0*res_mlp + coef1*gate*expert_out[e,slot,:]
// ---------------------------------------------------------------------------
__global__ __launch_bounds__(256)
void combine_kernel(const float* __restrict__ x,
                    const float* __restrict__ res_mlp,
                    const float* __restrict__ expert_out,
                    const int*   __restrict__ slot,
                    const int*   __restrict__ expert_id,
                    const float* __restrict__ gatep,
                    const float* __restrict__ c0,
                    const float* __restrict__ c1,
                    float* __restrict__ out) {
  const int s   = blockIdx.x;
  const int tid = threadIdx.x;
  const float a = c0[s];
  const int  sl = slot[s];
  const float g = (sl >= 0) ? (gatep[s] * c1[s]) : 0.f;
  const float* eo = (sl >= 0)
      ? expert_out + ((size_t)expert_id[s] * CAP + sl) * HIDDEN
      : nullptr;
#pragma unroll
  for (int j = 0; j < 4; ++j) {
    const int k = tid + 256 * j;
    const float moe = (sl >= 0) ? eo[k] : 0.f;
    out[(size_t)s * HIDDEN + k] =
        x[(size_t)s * HIDDEN + k] + a * res_mlp[(size_t)s * HIDDEN + k] + g * moe;
  }
}

// ---------------------------------------------------------------------------
extern "C" void kernel_launch(void* const* d_in, const int* in_sizes, int n_in,
                              void* d_out, int out_size, void* d_ws, size_t ws_size,
                              hipStream_t stream) {
  const float* x            = (const float*)d_in[0];
  const float* attn_nw      = (const float*)d_in[1];
  const float* attn_nb      = (const float*)d_in[2];
  const float* wg           = (const float*)d_in[3];
  const float* inter_w      = (const float*)d_in[4];
  const float* inter_b      = (const float*)d_in[5];
  const float* output_w     = (const float*)d_in[6];
  const float* output_b     = (const float*)d_in[7];
  const float* res_inter_w  = (const float*)d_in[8];
  const float* res_inter_b  = (const float*)d_in[9];
  const float* res_output_w = (const float*)d_in[10];
  const float* res_output_b = (const float*)d_in[11];
  const float* res_coef     = (const float*)d_in[12];
  float* out = (float*)d_out;

  char* ws = (char*)d_ws;
  size_t o = 0;
  auto alloc = [&](size_t bytes) { char* p = ws + o; o += (bytes + 255) & ~(size_t)255; return p; };

  unsigned short* attn_bf   = (unsigned short*)alloc((size_t)S_TOK * HIDDEN * 2);
  int*   expert_id          = (int*)  alloc((size_t)S_TOK * 4);
  float* gatep              = (float*)alloc((size_t)S_TOK * 4);
  float* coef0              = (float*)alloc((size_t)S_TOK * 4);
  float* coef1              = (float*)alloc((size_t)S_TOK * 4);
  int*   slot               = (int*)  alloc((size_t)S_TOK * 4);
  int*   tos                = (int*)  alloc((size_t)EXPERTS * CAP * 4);
  unsigned short* dispA     = (unsigned short*)alloc((size_t)EXPERTS * CAP * HIDDEN * 2);
  unsigned short* h         = (unsigned short*)alloc((size_t)EXPERTS * CAP * INTER * 2);
  unsigned short* rh        = (unsigned short*)alloc((size_t)S_TOK * INTER * 2);
  float* expert_out         = (float*)alloc((size_t)EXPERTS * CAP * HIDDEN * 4);
  float* res_mlp            = (float*)alloc((size_t)S_TOK * HIDDEN * 4);

  // 1) LayerNorm + gating
  ln_gate_kernel<<<S_TOK, 256, 0, stream>>>(x, attn_nw, attn_nb, wg, res_coef,
                                            attn_bf, expert_id, gatep, coef0, coef1);
  // 2) token-order capacity scan
  gate_scan_kernel<<<1, 1024, 0, stream>>>(expert_id, slot, tos);
  // 3) dispatch gather
  dispatch_kernel<<<EXPERTS * CAP, 128, 0, stream>>>(tos, attn_bf, dispA);

  // 4) expert GEMM1 + GELU: [E](CAP x HIDDEN) @ (HIDDEN x INTER) -> h (bf16)
  gemm_kernel<true, true><<<dim3(CAP / 128, INTER / 128, EXPERTS), 256, 0, stream>>>(
      dispA, (size_t)CAP * HIDDEN, inter_w, (size_t)HIDDEN * INTER,
      inter_b, (size_t)INTER, h, (size_t)CAP * INTER, INTER, HIDDEN);
  // 4b) residual GEMM1 + GELU: (S x HIDDEN) @ (HIDDEN x INTER) -> rh (bf16)
  gemm_kernel<true, true><<<dim3(S_TOK / 128, INTER / 128, 1), 256, 0, stream>>>(
      attn_bf, 0, res_inter_w, 0, res_inter_b, 0, rh, 0, INTER, HIDDEN);

  // 5) expert GEMM2: [E](CAP x INTER) @ (INTER x HIDDEN) -> expert_out (f32)
  gemm_kernel<false, false><<<dim3(CAP / 128, HIDDEN / 128, EXPERTS), 256, 0, stream>>>(
      h, (size_t)CAP * INTER, output_w, (size_t)INTER * HIDDEN,
      output_b, (size_t)HIDDEN, expert_out, (size_t)CAP * HIDDEN, HIDDEN, INTER);
  // 5b) residual GEMM2: (S x INTER) @ (INTER x HIDDEN) -> res_mlp (f32)
  gemm_kernel<false, false><<<dim3(S_TOK / 128, HIDDEN / 128, 1), 256, 0, stream>>>(
      rh, 0, res_output_w, 0, res_output_b, 0, res_mlp, 0, HIDDEN, INTER);

  // 6) gated combine + residual add
  combine_kernel<<<S_TOK, 256, 0, stream>>>(x, res_mlp, expert_out, slot,
                                            expert_id, gatep, coef0, coef1, out);
}
